// GSN_sparse_58737972740096
// MI455X (gfx1250) — matrix operand order repaired and checked by reading, not compiled, not run on hardware
//
#include <hip/hip_runtime.h>
#include <hip/hip_bf16.h>

#define NN 50000
#define NE 800000
#define ET 5   // edge tiles per wave: 50000 / (8*5) = 1250 blocks exact
#define NT 5   // node tiles per wave

typedef __attribute__((ext_vector_type(16))) __bf16 v16bf;
typedef __attribute__((ext_vector_type(8)))  float  v8f;

struct F32B { int4 lo; int4 hi; };

static __device__ inline unsigned short f2bf(float f) {
    unsigned int u = __float_as_uint(f);
    u = (u + 0x7FFFu + ((u >> 16) & 1u)) >> 16;   // round-to-nearest-even
    return (unsigned short)u;
}

static __device__ inline v16bf frag_from(const void* p) {
    F32B f;
    f.lo = ((const int4*)p)[0];
    f.hi = ((const int4*)p)[1];
    return __builtin_bit_cast(v16bf, f);
}

static __device__ inline unsigned lds_off(const void* p) {
    // flat address of a __shared__ object: low 32 bits == LDS byte offset
    return (unsigned)(unsigned long long)p;
}

// async global -> LDS (CDNA5): vdst = LDS byte addr, vaddr = 32-bit byte
// offset, saddr = 64-bit base (GVS mode). Tracked with ASYNCcnt.
static __device__ inline void async_g2l_b128(unsigned lds, unsigned voff, const void* base) {
    asm volatile("global_load_async_to_lds_b128 %0, %1, %2"
                 :: "v"(lds), "v"(voff), "s"(base) : "memory");
}
static __device__ inline void async_g2l_b32(unsigned lds, unsigned voff, const void* base) {
    asm volatile("global_load_async_to_lds_b32 %0, %1, %2"
                 :: "v"(lds), "v"(voff), "s"(base) : "memory");
}
static __device__ inline void wait_async0() {
    asm volatile("s_wait_asynccnt 0x0" ::: "memory");
}

// ---------------- prep kernels ----------------

__global__ void k_zero(float* __restrict__ p, int n) {
    int i = blockIdx.x * blockDim.x + threadIdx.x;
    if (i < n) p[i] = 0.0f;
}

__global__ void k_cvt(const float* __restrict__ s, unsigned short* __restrict__ d, int n) {
    int i = blockIdx.x * blockDim.x + threadIdx.x;
    if (i < n) d[i] = f2bf(s[i]);
}

// Pre-swizzle a [Ktot x Ntot] row-major f32 weight matrix into bf16 WMMA
// B-fragments: frag f = kt*NTt+nt, lane l holds n = nt*16 + l%16 and
// k = kt*32 + (l/16)*16 + e (e = 0..15), stored lane-contiguously (32 B/lane).
__global__ void k_swz(const float* __restrict__ W, unsigned short* __restrict__ o,
                      int Ktot, int Ntot) {
    int t = blockIdx.x * blockDim.x + threadIdx.x;
    if (t >= Ktot * Ntot) return;
    int e   = t & 15;
    int l   = (t >> 4) & 31;
    int f   = t >> 9;
    int NTt = Ntot >> 4;
    int kt  = f / NTt, nt = f - kt * NTt;
    int k   = kt * 32 + (l >> 4) * 16 + e;
    int n   = nt * 16 + (l & 15);
    o[t] = f2bf(W[k * Ntot + n]);
}

// ---------------- edge kernel: fused gather + MLP + scatter-add ----------------
// 1 wave = ET tiles of 16 edges. grid = NE/16/(8*ET) blocks of 256 (exact).

__global__ __launch_bounds__(256) void edge_kernel(
    const int* __restrict__ eidx,            // [2][NE] (int32)
    const unsigned short* __restrict__ xb,   // [NN][64] bf16
    const unsigned short* __restrict__ idb,  // [NN][32] bf16
    const float* __restrict__ bm1, const float* __restrict__ bm2,
    const unsigned short* __restrict__ wm1f, // swizzled 192x128
    const unsigned short* __restrict__ wm2f, // swizzled 128x64
    float* __restrict__ agg)                 // [NN][64] f32
{
    __shared__ unsigned int sW1[12288];      // 192*128 bf16 = 48 KB
    __shared__ unsigned int sW2[4096];       // 128*64  bf16 = 16 KB
    __shared__ unsigned int sIn[8][1600];    // per-wave 16 x (192+8 pad) bf16
    __shared__ unsigned int sH [8][1088];    // per-wave 16 x (128+8 pad) bf16

    const int tid  = threadIdx.x;
    const int wave = tid >> 5;
    const int lane = tid & 31;
    const int half = lane >> 4;
    const int l16  = lane & 15;

    {   // stage weights block-wide via async copies (16B chunks)
        unsigned l1 = lds_off(sW1);
        for (int i = tid; i < 3072; i += 256) async_g2l_b128(l1 + i * 16, i * 16, wm1f);
        unsigned l2 = lds_off(sW2);
        for (int i = tid; i < 1024; i += 256) async_g2l_b128(l2 + i * 16, i * 16, wm2f);
    }
    wait_async0();
    __syncthreads();

    unsigned int*   myIn = sIn[wave];
    unsigned short* myH  = (unsigned short*)sH[wave];
    const unsigned  inB  = lds_off(myIn);

    for (int t = 0; t < ET; ++t) {
        const int tile  = blockIdx.x * (8 * ET) + wave * ET + t;
        const int ebase = tile * 16;

        // async-gather [16 x 192] bf16 m_in tile = [x_i | x_j | id_i | id_j]
        for (int m = 0; m < 16; ++m) {
            int nj = eidx[ebase + m];        // source j
            int ni = eidx[NE + ebase + m];   // target i
            async_g2l_b32(inB + (m * 100 + lane) * 4,
                          (unsigned)(ni * 128 + lane * 4), xb);
            async_g2l_b32(inB + (m * 100 + 32 + lane) * 4,
                          (unsigned)(nj * 128 + lane * 4), xb);
            unsigned io = (lane < 16) ? (unsigned)(ni * 64 + lane * 4)
                                      : (unsigned)(nj * 64 + (lane - 16) * 4);
            async_g2l_b32(inB + (m * 100 + 64 + lane) * 4, io, idb);
        }
        wait_async0();
        __syncthreads();

        // ---- layer 1: [16x192] @ [192x128] + b, ReLU ----
        v16bf a[6];
        {
            const char* base = (const char*)myIn + l16 * 400 + half * 32;
            #pragma unroll
            for (int kt = 0; kt < 6; ++kt) a[kt] = frag_from(base + kt * 64);
        }
        #pragma unroll
        for (int nt = 0; nt < 8; ++nt) {
            float bv = bm1[nt * 16 + l16];
            v8f c;
            #pragma unroll
            for (int r = 0; r < 8; ++r) c[r] = bv;
            #pragma unroll
            for (int kt = 0; kt < 6; ++kt) {
                v16bf b = frag_from((const char*)sW1 + ((kt * 8 + nt) * 32 + lane) * 32);
                c = __builtin_amdgcn_wmma_f32_16x16x32_bf16(false, a[kt], false, b,
                                                            (short)0, c, false, false);
            }
            #pragma unroll
            for (int r = 0; r < 8; ++r) {
                float v = c[r] > 0.0f ? c[r] : 0.0f;
                myH[(half * 8 + r) * 136 + nt * 16 + l16] = f2bf(v);
            }
        }
        __syncthreads();

        // ---- layer 2: [16x128] @ [128x64] + b, then scatter-add to agg ----
        v16bf a2[4];
        {
            const char* base = (const char*)myH + l16 * 272 + half * 32;
            #pragma unroll
            for (int kt = 0; kt < 4; ++kt) a2[kt] = frag_from(base + kt * 64);
        }
        int tgt[8];
        #pragma unroll
        for (int r = 0; r < 8; ++r) tgt[r] = eidx[NE + ebase + half * 8 + r];

        #pragma unroll
        for (int nt = 0; nt < 4; ++nt) {
            float bv = bm2[nt * 16 + l16];
            v8f c;
            #pragma unroll
            for (int r = 0; r < 8; ++r) c[r] = bv;
            #pragma unroll
            for (int kt = 0; kt < 4; ++kt) {
                v16bf b = frag_from((const char*)sW2 + ((kt * 4 + nt) * 32 + lane) * 32);
                c = __builtin_amdgcn_wmma_f32_16x16x32_bf16(false, a2[kt], false, b,
                                                            (short)0, c, false, false);
            }
            #pragma unroll
            for (int r = 0; r < 8; ++r)
                unsafeAtomicAdd(&agg[tgt[r] * 64 + nt * 16 + l16], c[r]);
        }
    }
}

// ---------------- node kernel: [x | agg] MLP -> out ----------------
// 1 wave = NT tiles of 16 nodes; tail waves clamp (duplicate identical stores).

__global__ __launch_bounds__(256) void node_kernel(
    const unsigned short* __restrict__ xb,
    const float* __restrict__ agg,
    const float* __restrict__ bu1, const float* __restrict__ bu2,
    const unsigned short* __restrict__ wu1f,  // swizzled 128x128
    const unsigned short* __restrict__ wu2f,  // swizzled 128x64
    float* __restrict__ out)                  // [NN][64]
{
    __shared__ unsigned int sW1[8192];        // 32 KB
    __shared__ unsigned int sW2[4096];        // 16 KB
    __shared__ unsigned int sIn[8][1088];
    __shared__ unsigned int sH [8][1088];

    const int tid  = threadIdx.x;
    const int wave = tid >> 5;
    const int lane = tid & 31;
    const int half = lane >> 4;
    const int l16  = lane & 15;

    {
        unsigned l1 = lds_off(sW1);
        for (int i = tid; i < 2048; i += 256) async_g2l_b128(l1 + i * 16, i * 16, wu1f);
        unsigned l2 = lds_off(sW2);
        for (int i = tid; i < 1024; i += 256) async_g2l_b128(l2 + i * 16, i * 16, wu2f);
    }
    wait_async0();
    __syncthreads();

    unsigned int*   myIn = sIn[wave];
    unsigned short* myH  = (unsigned short*)sH[wave];
    const unsigned  inB  = lds_off(myIn);
    const int ntiles = NN / 16;   // 3125

    for (int t = 0; t < NT; ++t) {
        int tile = blockIdx.x * (8 * NT) + wave * NT + t;
        if (tile >= ntiles) tile = ntiles - 1;
        const int nbase = tile * 16;

        for (int m = 0; m < 16; ++m) {
            int node = nbase + m;
            async_g2l_b32(inB + (m * 68 + lane) * 4,
                          (unsigned)(node * 128 + lane * 4), xb);
            float a0 = agg[node * 64 + 2 * lane];
            float a1 = agg[node * 64 + 2 * lane + 1];
            myIn[m * 68 + 32 + lane] =
                (unsigned int)f2bf(a0) | ((unsigned int)f2bf(a1) << 16);
        }
        wait_async0();
        __syncthreads();

        // ---- layer 1: [16x128] @ [128x128] + b, ReLU ----
        v16bf a[4];
        {
            const char* base = (const char*)myIn + l16 * 272 + half * 32;
            #pragma unroll
            for (int kt = 0; kt < 4; ++kt) a[kt] = frag_from(base + kt * 64);
        }
        #pragma unroll
        for (int nt = 0; nt < 8; ++nt) {
            float bv = bu1[nt * 16 + l16];
            v8f c;
            #pragma unroll
            for (int r = 0; r < 8; ++r) c[r] = bv;
            #pragma unroll
            for (int kt = 0; kt < 4; ++kt) {
                v16bf b = frag_from((const char*)sW1 + ((kt * 8 + nt) * 32 + lane) * 32);
                c = __builtin_amdgcn_wmma_f32_16x16x32_bf16(false, a[kt], false, b,
                                                            (short)0, c, false, false);
            }
            #pragma unroll
            for (int r = 0; r < 8; ++r) {
                float v = c[r] > 0.0f ? c[r] : 0.0f;
                myH[(half * 8 + r) * 136 + nt * 16 + l16] = f2bf(v);
            }
        }
        __syncthreads();

        // ---- layer 2: [16x128] @ [128x64] + b -> out ----
        v16bf a2[4];
        {
            const char* base = (const char*)myH + l16 * 272 + half * 32;
            #pragma unroll
            for (int kt = 0; kt < 4; ++kt) a2[kt] = frag_from(base + kt * 64);
        }
        #pragma unroll
        for (int nt = 0; nt < 4; ++nt) {
            float bv = bu2[nt * 16 + l16];
            v8f c;
            #pragma unroll
            for (int r = 0; r < 8; ++r) c[r] = bv;
            #pragma unroll
            for (int kt = 0; kt < 4; ++kt) {
                v16bf b = frag_from((const char*)sW2 + ((kt * 4 + nt) * 32 + lane) * 32);
                c = __builtin_amdgcn_wmma_f32_16x16x32_bf16(false, a2[kt], false, b,
                                                            (short)0, c, false, false);
            }
            #pragma unroll
            for (int r = 0; r < 8; ++r)
                out[(nbase + half * 8 + r) * 64 + nt * 16 + l16] = c[r];
        }
    }
}

// ---------------- launch ----------------

extern "C" void kernel_launch(void* const* d_in, const int* in_sizes, int n_in,
                              void* d_out, int out_size, void* d_ws, size_t ws_size,
                              hipStream_t stream) {
    const float* x   = (const float*)d_in[0];
    const float* ids = (const float*)d_in[1];
    // d_in[2] = degrees (unused by reference)
    const int*   eidx = (const int*)d_in[3];
    const float* Wm1 = (const float*)d_in[4];
    const float* bm1 = (const float*)d_in[5];
    const float* Wm2 = (const float*)d_in[6];
    const float* bm2 = (const float*)d_in[7];
    const float* Wu1 = (const float*)d_in[8];
    const float* bu1 = (const float*)d_in[9];
    const float* Wu2 = (const float*)d_in[10];
    const float* bu2 = (const float*)d_in[11];

    char* ws = (char*)d_ws;
    float*          agg  = (float*)ws;                              // 12,800,000 B
    unsigned short* xb   = (unsigned short*)(ws + 12800000);        //  6,400,000 B
    unsigned short* idb  = (unsigned short*)(ws + 19200000);        //  3,200,000 B
    unsigned short* wm1f = (unsigned short*)(ws + 22400000);        //     49,152 B
    unsigned short* wm2f = (unsigned short*)(ws + 22449152);        //     16,384 B
    unsigned short* wu1f = (unsigned short*)(ws + 22465536);        //     32,768 B
    unsigned short* wu2f = (unsigned short*)(ws + 22498304);        //     16,384 B

    k_zero<<<(NN * 64 + 255) / 256, 256, 0, stream>>>(agg, NN * 64);
    k_cvt <<<(NN * 64 + 255) / 256, 256, 0, stream>>>(x,   xb,  NN * 64);
    k_cvt <<<(NN * 32 + 255) / 256, 256, 0, stream>>>(ids, idb, NN * 32);
    k_swz <<<(192 * 128 + 255) / 256, 256, 0, stream>>>(Wm1, wm1f, 192, 128);
    k_swz <<<(128 *  64 + 255) / 256, 256, 0, stream>>>(Wm2, wm2f, 128,  64);
    k_swz <<<(128 * 128 + 255) / 256, 256, 0, stream>>>(Wu1, wu1f, 128, 128);
    k_swz <<<(128 *  64 + 255) / 256, 256, 0, stream>>>(Wu2, wu2f, 128,  64);

    edge_kernel<<<NE / 16 / (8 * ET), 256, 0, stream>>>(eidx, xb, idb, bm1, bm2,
                                                        wm1f, wm2f, agg);
    node_kernel<<<(NN / 16 + 8 * NT - 1) / (8 * NT), 256, 0, stream>>>(
        xb, agg, bu1, bu2, wu1f, wu2f, (float*)d_out);
}
